// ParallelGroupedMergedMLP_9294309228785
// MI455X (gfx1250) — matrix operand order repaired and checked by reading, not compiled, not run on hardware
//
#include <hip/hip_runtime.h>
#include <hip/hip_bf16.h>

// ---------------------------------------------------------------------------
// MoE grouped merged MLP for MI455X (gfx1250, wave32, WMMA).
//   merged_w{1,3}[g] = sum_e ew[g,e]*w{1,3}[e]   (stored bf16, transposed N-major)
//   merged_w2[g]     = sum_e ew[g,e]*w2[e]       (stored bf16, transposed N-major)
//   T[g]   = silu(X[g]·W1[g]) * (X[g]·W3[g])     (bf16 intermediate)
//   out[g] = T[g]·W2[g]                          (fp32, scattered back to [S,G,H])
// Memory-bound: weights read once in fp32, re-read in bf16 -> ~2.6 GB total.
// Workspace requirement: ~752 MB.
// GEMMs use double-buffered GLOBAL_LOAD_ASYNC_TO_LDS (ASYNCcnt) when the
// toolchain exposes the builtin; otherwise fall back to sync VGPR staging.
// ---------------------------------------------------------------------------

#define HDIM 2048
#define EXP  8
#define IDIM 5461
#define SEQ  2048
#define GRP  8
#define IPAD 5504   // 43 * 128, zero-padded columns

#define BM 128
#define BN 128
#define BK 32

typedef unsigned short u16;
typedef __attribute__((ext_vector_type(16))) __bf16 v16bf;
typedef __attribute__((ext_vector_type(8)))  float  v8f;

union Frag16 { v16bf v; uint4 q[2]; };

// ---- async global->LDS support (guarded; falls back to sync staging) ------
#if defined(__has_builtin)
#if __has_builtin(__builtin_amdgcn_global_load_async_to_lds_b128)
#define USE_ASYNC_LDS 1
#endif
#endif
#ifndef USE_ASYNC_LDS
#define USE_ASYNC_LDS 0
#endif

#if USE_ASYNC_LDS
// builtin expects true clang vector-of-4-int pointers, addrspace-qualified:
//   param0: v4i addrspace(1)* (global source), param1: v4i addrspace(3)* (LDS dest)
typedef int v4i __attribute__((vector_size(16)));
typedef __attribute__((address_space(1))) v4i GV4i;
typedef __attribute__((address_space(3))) v4i LV4i;
__device__ __forceinline__ void async_cp16(void* lds, const void* g) {
    // per-lane 16B copy from global to LDS, tracked by ASYNCcnt
    __builtin_amdgcn_global_load_async_to_lds_b128((GV4i*)(g), (LV4i*)(lds), 0, 0);
}
#if __has_builtin(__builtin_amdgcn_s_wait_asynccnt)
#define WAIT_ASYNC(n) __builtin_amdgcn_s_wait_asynccnt(n)
#else
#define WAIT_ASYNC(n) asm volatile("s_wait_asynccnt %0" ::"i"(n) : "memory")
#endif
#endif // USE_ASYNC_LDS

__device__ __forceinline__ u16 f2bf(float f) {
    union { float f; unsigned u; } cvt; cvt.f = f;
    unsigned u = cvt.u;
    unsigned r = u + 0x7FFFu + ((u >> 16) & 1u);   // round-to-nearest-even
    return (u16)(r >> 16);
}

// ---------------------------------------------------------------------------
// Kernel 1: hidden_states [S,G,H] fp32 -> Xb [G][S][H] bf16
// ---------------------------------------------------------------------------
__global__ __launch_bounds__(256) void moe_convert_x(
    const float* __restrict__ hs, u16* __restrict__ Xb)
{
    size_t idx = ((size_t)blockIdx.x * blockDim.x + threadIdx.x) * 4; // [G][S][H] flat
    int h = (int)(idx % HDIM);
    size_t t = idx / HDIM;
    int s = (int)(t % SEQ);
    int g = (int)(t / SEQ);
    const float4 v = *(const float4*)(hs + ((size_t)s * GRP + g) * HDIM + h);
    ushort4 o;
    o.x = f2bf(v.x); o.y = f2bf(v.y); o.z = f2bf(v.z); o.w = f2bf(v.w);
    *(ushort4*)(Xb + idx) = o;
}

// ---------------------------------------------------------------------------
// Kernel 2: merge over experts + transpose + bf16 quantize.
//   out[g][n][k] = bf16( sum_e ew[g,e] * w[e][k][n] ),  zero for k>=K or n>=N.
//   w is [E][K][N] fp32, out is [G][Npad][Kpad] bf16.
// One 32x32 (k,n) tile per block, all G groups handled (weights read ONCE).
// ---------------------------------------------------------------------------
__global__ __launch_bounds__(256) void moe_merge_transpose(
    const float* __restrict__ w, const float* __restrict__ ew,
    u16* __restrict__ out, int K, int N, int Kpad, int Npad)
{
    const int k0 = blockIdx.x * 32;
    const int n0 = blockIdx.y * 32;
    const int tid = threadIdx.x;
    const int tx = tid & 31;     // along n (reads) / along k (writes)
    const int ty = tid >> 5;     // 0..7

    __shared__ float sEw[GRP * EXP];
    __shared__ float tile[32][33];

    if (tid < GRP * EXP) sEw[tid] = ew[tid];
    __syncthreads();

    float acc[4][GRP];
#pragma unroll
    for (int r = 0; r < 4; ++r)
#pragma unroll
        for (int g = 0; g < GRP; ++g) acc[r][g] = 0.f;

    const int n = n0 + tx;
#pragma unroll
    for (int e = 0; e < EXP; ++e) {
#pragma unroll
        for (int r = 0; r < 4; ++r) {
            const int k = k0 + ty + r * 8;
            float val = 0.f;
            if (k < K && n < N)
                val = w[((size_t)e * K + k) * N + n];
#pragma unroll
            for (int g = 0; g < GRP; ++g)
                acc[r][g] = fmaf(sEw[g * EXP + e], val, acc[r][g]);
        }
    }

    for (int g = 0; g < GRP; ++g) {
        __syncthreads();
#pragma unroll
        for (int r = 0; r < 4; ++r)
            tile[ty + r * 8][tx] = acc[r][g];   // tile[k_rel][n_rel]
        __syncthreads();
#pragma unroll
        for (int r = 0; r < 4; ++r) {
            const int nr = ty + r * 8;
            out[((size_t)g * Npad + (n0 + nr)) * Kpad + (k0 + tx)] =
                f2bf(tile[tx][nr]);             // coalesced along k
        }
    }
}

// ---------------------------------------------------------------------------
// GEMM fragment helpers (wave32 WMMA 16x16x32 bf16 layouts)
//   A (16x32, 16-bit): lane l (row=l&15, half=l>>4) holds K in
//       [8*half, 8*half+8) and [16+8*half, 16+8*half+8)  -> 2x 16B LDS loads.
//   B (32x16, 16-bit, LDS stored N-major [n][k]): lane l (col=l&15) holds K in
//       [8*half, 8*half+8) and [16+8*half, 16+8*half+8)  -> 2x 16B LDS loads.
// ---------------------------------------------------------------------------
__device__ __forceinline__ Frag16 load_fragA(const u16* lA, int row, int half) {
    Frag16 f;
    const u16* p = lA + row * BK + 8 * half;
    f.q[0] = *(const uint4*)(p);
    f.q[1] = *(const uint4*)(p + 16);
    return f;
}
__device__ __forceinline__ Frag16 load_fragB(const u16* lB, int col, int half) {
    Frag16 f;
    const u16* p = lB + col * BK + 8 * half;
    f.q[0] = *(const uint4*)(p);
    f.q[1] = *(const uint4*)(p + 16);
    return f;
}

// ---------------------------------------------------------------------------
// Kernel 3: fused gated GEMM.  For group g, tile (m0,n0):
//   T = silu(X·W1t^T) * (X·W3t^T)   with X [S,H] bf16, W1t/W3t [IPAD,H] bf16.
// ---------------------------------------------------------------------------
__global__ __launch_bounds__(256) void moe_gemm1_gate(
    const u16* __restrict__ Xb, const u16* __restrict__ W1t,
    const u16* __restrict__ W3t, u16* __restrict__ T)
{
    const int g  = blockIdx.z;
    const int m0 = blockIdx.y * BM;
    const int n0 = blockIdx.x * BN;
    const int tid   = threadIdx.x;
    const int lane  = tid & 31;
    const int wave  = tid >> 5;
    const int waveM = (wave >> 2) * 64;   // 0, 64
    const int waveN = (wave & 3) * 32;    // 0..96
    const int half  = lane >> 4;
    const int l16   = lane & 15;

    const u16* Ag  = Xb  + ((size_t)g * SEQ  + m0) * HDIM;
    const u16* B1g = W1t + ((size_t)g * IPAD + n0) * HDIM;
    const u16* B3g = W3t + ((size_t)g * IPAD + n0) * HDIM;

    v8f acc1[4][2], acc3[4][2];
#pragma unroll
    for (int mt = 0; mt < 4; ++mt)
#pragma unroll
        for (int nt = 0; nt < 2; ++nt) { acc1[mt][nt] = {}; acc3[mt][nt] = {}; }

#if USE_ASYNC_LDS
    __shared__ u16 lA [2][BM * BK];
    __shared__ u16 lB1[2][BN * BK];
    __shared__ u16 lB3[2][BN * BK];

    // 6 async b128 copies per thread per K-slab (2 each for A, B1, B3)
    auto stage = [&](int buf, int kb) {
#pragma unroll
        for (int i = 0; i < 2; ++i) {
            const int idx = tid + i * 256;
            const int r = idx >> 2, c = idx & 3;
            async_cp16(&lA [buf][idx * 8], Ag  + (size_t)r * HDIM + kb + c * 8);
            async_cp16(&lB1[buf][idx * 8], B1g + (size_t)r * HDIM + kb + c * 8);
            async_cp16(&lB3[buf][idx * 8], B3g + (size_t)r * HDIM + kb + c * 8);
        }
    };

    stage(0, 0);
    int cur = 0;
    for (int kb = 0; kb < HDIM; kb += BK) {
        if (kb + BK < HDIM) { stage(cur ^ 1, kb + BK); WAIT_ASYNC(6); }
        else                { WAIT_ASYNC(0); }
        __syncthreads();                       // all waves' copies for cur landed
        const u16* a  = lA [cur];
        const u16* b1 = lB1[cur];
        const u16* b3 = lB3[cur];
#else
    __shared__ u16 lA [BM * BK];
    __shared__ u16 lB1[BN * BK];
    __shared__ u16 lB3[BN * BK];
    for (int kb = 0; kb < HDIM; kb += BK) {
#pragma unroll
        for (int i = 0; i < 2; ++i) {
            const int idx = tid + i * 256;          // 0..511
            const int r = idx >> 2, c = idx & 3;    // 8 bf16 per chunk
            ((uint4*)lA )[idx] = *(const uint4*)(Ag  + (size_t)r * HDIM + kb + c * 8);
            ((uint4*)lB1)[idx] = *(const uint4*)(B1g + (size_t)r * HDIM + kb + c * 8);
            ((uint4*)lB3)[idx] = *(const uint4*)(B3g + (size_t)r * HDIM + kb + c * 8);
        }
        if (kb + BK < HDIM) {   // global_prefetch_b8 of next K-slab
            __builtin_prefetch(Ag  + (size_t)(tid >> 1) * HDIM + kb + BK + (tid & 1) * 16, 0, 1);
            __builtin_prefetch(B1g + (size_t)(tid >> 1) * HDIM + kb + BK + (tid & 1) * 16, 0, 1);
            __builtin_prefetch(B3g + (size_t)(tid >> 1) * HDIM + kb + BK + (tid & 1) * 16, 0, 1);
        }
        __syncthreads();
        const u16* a  = lA;
        const u16* b1 = lB1;
        const u16* b3 = lB3;
#endif
        Frag16 af[4], bf1[2], bf3[2];
#pragma unroll
        for (int mt = 0; mt < 4; ++mt)
            af[mt] = load_fragA(a, waveM + mt * 16 + l16, half);
#pragma unroll
        for (int nt = 0; nt < 2; ++nt) {
            bf1[nt] = load_fragB(b1, waveN + nt * 16 + l16, half);
            bf3[nt] = load_fragB(b3, waveN + nt * 16 + l16, half);
        }
#pragma unroll
        for (int mt = 0; mt < 4; ++mt)
#pragma unroll
            for (int nt = 0; nt < 2; ++nt) {
                acc1[mt][nt] = __builtin_amdgcn_wmma_f32_16x16x32_bf16(
                    false, af[mt].v, false, bf1[nt].v, (short)0, acc1[mt][nt], false, false);
                acc3[mt][nt] = __builtin_amdgcn_wmma_f32_16x16x32_bf16(
                    false, af[mt].v, false, bf3[nt].v, (short)0, acc3[mt][nt], false, false);
            }
        __syncthreads();                       // reads done; buffer may be rewritten
#if USE_ASYNC_LDS
        cur ^= 1;
#endif
    }

    // Epilogue: t = silu(o1) * o3, store bf16. C/D layout: VGPR r ->
    // (row = r + 8*half, col = lane&15) within each 16x16 tile.
    u16* Tg = T + (size_t)g * SEQ * IPAD;
#pragma unroll
    for (int mt = 0; mt < 4; ++mt)
#pragma unroll
        for (int nt = 0; nt < 2; ++nt)
#pragma unroll
            for (int r = 0; r < 8; ++r) {
                const int srow = m0 + waveM + mt * 16 + r + 8 * half;
                const int icol = n0 + waveN + nt * 16 + l16;
                const float o1 = acc1[mt][nt][r];
                const float o3 = acc3[mt][nt][r];
                const float t  = (o1 / (1.f + __expf(-o1))) * o3;
                Tg[(size_t)srow * IPAD + icol] = f2bf(t);
            }
}

// ---------------------------------------------------------------------------
// Kernel 4: down-projection GEMM. out[s,g,h] = T[g]·W2t[g]^T (fp32 out).
//   T [S,IPAD] bf16, W2t [H,IPAD] bf16 (padded K contributes zero).
// ---------------------------------------------------------------------------
__global__ __launch_bounds__(256) void moe_gemm2_down(
    const u16* __restrict__ T, const u16* __restrict__ W2t,
    float* __restrict__ out)
{
    const int g  = blockIdx.z;
    const int m0 = blockIdx.y * BM;
    const int n0 = blockIdx.x * BN;
    const int tid   = threadIdx.x;
    const int lane  = tid & 31;
    const int wave  = tid >> 5;
    const int waveM = (wave >> 2) * 64;
    const int waveN = (wave & 3) * 32;
    const int half  = lane >> 4;
    const int l16   = lane & 15;

    const u16* Ag = T   + ((size_t)g * SEQ  + m0) * IPAD;
    const u16* Bg = W2t + ((size_t)g * HDIM + n0) * IPAD;

    v8f acc[4][2];
#pragma unroll
    for (int mt = 0; mt < 4; ++mt)
#pragma unroll
        for (int nt = 0; nt < 2; ++nt) acc[mt][nt] = {};

#if USE_ASYNC_LDS
    __shared__ u16 lA[2][BM * BK];
    __shared__ u16 lB[2][BN * BK];

    auto stage = [&](int buf, int kb) {   // 4 async b128 copies per thread
#pragma unroll
        for (int i = 0; i < 2; ++i) {
            const int idx = tid + i * 256;
            const int r = idx >> 2, c = idx & 3;
            async_cp16(&lA[buf][idx * 8], Ag + (size_t)r * IPAD + kb + c * 8);
            async_cp16(&lB[buf][idx * 8], Bg + (size_t)r * IPAD + kb + c * 8);
        }
    };

    stage(0, 0);
    int cur = 0;
    for (int kb = 0; kb < IPAD; kb += BK) {
        if (kb + BK < IPAD) { stage(cur ^ 1, kb + BK); WAIT_ASYNC(4); }
        else                { WAIT_ASYNC(0); }
        __syncthreads();
        const u16* a = lA[cur];
        const u16* b = lB[cur];
#else
    __shared__ u16 lA[BM * BK];
    __shared__ u16 lB[BN * BK];
    for (int kb = 0; kb < IPAD; kb += BK) {
#pragma unroll
        for (int i = 0; i < 2; ++i) {
            const int idx = tid + i * 256;
            const int r = idx >> 2, c = idx & 3;
            ((uint4*)lA)[idx] = *(const uint4*)(Ag + (size_t)r * IPAD + kb + c * 8);
            ((uint4*)lB)[idx] = *(const uint4*)(Bg + (size_t)r * IPAD + kb + c * 8);
        }
        if (kb + BK < IPAD) {
            __builtin_prefetch(Ag + (size_t)(tid >> 1) * IPAD + kb + BK + (tid & 1) * 16, 0, 1);
            __builtin_prefetch(Bg + (size_t)(tid >> 1) * IPAD + kb + BK + (tid & 1) * 16, 0, 1);
        }
        __syncthreads();
        const u16* a = lA;
        const u16* b = lB;
#endif
        Frag16 af[4], bf[2];
#pragma unroll
        for (int mt = 0; mt < 4; ++mt)
            af[mt] = load_fragA(a, waveM + mt * 16 + l16, half);
#pragma unroll
        for (int nt = 0; nt < 2; ++nt)
            bf[nt] = load_fragB(b, waveN + nt * 16 + l16, half);
#pragma unroll
        for (int mt = 0; mt < 4; ++mt)
#pragma unroll
            for (int nt = 0; nt < 2; ++nt)
                acc[mt][nt] = __builtin_amdgcn_wmma_f32_16x16x32_bf16(
                    false, af[mt].v, false, bf[nt].v, (short)0, acc[mt][nt], false, false);
        __syncthreads();
#if USE_ASYNC_LDS
        cur ^= 1;
#endif
    }

#pragma unroll
    for (int mt = 0; mt < 4; ++mt)
#pragma unroll
        for (int nt = 0; nt < 2; ++nt)
#pragma unroll
            for (int r = 0; r < 8; ++r) {
                const int srow = m0 + waveM + mt * 16 + r + 8 * half;
                const int hcol = n0 + waveN + nt * 16 + l16;
                out[((size_t)srow * GRP + g) * HDIM + hcol] = acc[mt][nt][r];
            }
}

// ---------------------------------------------------------------------------
extern "C" void kernel_launch(void* const* d_in, const int* in_sizes, int n_in,
                              void* d_out, int out_size, void* d_ws, size_t ws_size,
                              hipStream_t stream) {
    const float* hidden = (const float*)d_in[0];   // [S,G,H]
    const float* ew     = (const float*)d_in[1];   // [G,E]
    const float* w1     = (const float*)d_in[2];   // [E,H,I]
    const float* w2     = (const float*)d_in[3];   // [E,I,H]
    const float* w3     = (const float*)d_in[4];   // [E,H,I]
    float* out          = (float*)d_out;           // [S,G,H]

    char* ws = (char*)d_ws;
    const size_t xb_bytes = (size_t)GRP * SEQ  * HDIM * sizeof(u16); //  64 MiB
    const size_t wt_bytes = (size_t)GRP * IPAD * HDIM * sizeof(u16); // 172 MiB
    u16* Xb  = (u16*)(ws);
    u16* W1t = (u16*)(ws + xb_bytes);                                // [G][IPAD][H]
    u16* W3t = (u16*)(ws + xb_bytes + wt_bytes);                     // [G][IPAD][H]
    u16* W2t = (u16*)(ws + xb_bytes + 2 * wt_bytes);                 // [G][H][IPAD]
    u16* T   = (u16*)(ws + xb_bytes + 3 * wt_bytes);                 // [G][S][IPAD]
    (void)in_sizes; (void)n_in; (void)out_size; (void)ws_size;

    // 1) activations -> bf16, [S,G,H] -> [G][S][H]
    moe_convert_x<<<(GRP * SEQ * HDIM / 4) / 256, 256, 0, stream>>>(hidden, Xb);

    // 2) expert merge + transpose + bf16 (weights streamed once)
    moe_merge_transpose<<<dim3(HDIM / 32, IPAD / 32), 256, 0, stream>>>(
        w1, ew, W1t, /*K=*/HDIM, /*N=*/IDIM, /*Kpad=*/HDIM, /*Npad=*/IPAD);
    moe_merge_transpose<<<dim3(HDIM / 32, IPAD / 32), 256, 0, stream>>>(
        w3, ew, W3t, /*K=*/HDIM, /*N=*/IDIM, /*Kpad=*/HDIM, /*Npad=*/IPAD);
    moe_merge_transpose<<<dim3(IPAD / 32, HDIM / 32), 256, 0, stream>>>(
        w2, ew, W2t, /*K=*/IDIM, /*N=*/HDIM, /*Kpad=*/IPAD, /*Npad=*/HDIM);

    // 3) fused gated GEMM -> T
    moe_gemm1_gate<<<dim3(IPAD / BN, SEQ / BM, GRP), 256, 0, stream>>>(Xb, W1t, W3t, T);

    // 4) down projection -> out [S,G,H]
    moe_gemm2_down<<<dim3(HDIM / BN, SEQ / BM, GRP), 256, 0, stream>>>(T, W2t, out);
}
